// NodewiseInteraction_36112085025169
// MI455X (gfx1250) — matrix operand order repaired and checked by reading, not compiled, not run on hardware
//
#include <hip/hip_runtime.h>
#include <hip/hip_bf16.h>
#include <math.h>

// ---------------- problem constants ----------------
#define N_NODES 8192
#define N_EDGES 131072
#define NS 128
#define NV 64
#define DFEAT 320          // NS + 3*NV
#define HID 128
#define WNUM 384           // NS+NS+NV+NV
#define EPSLN 1e-5f

typedef __attribute__((ext_vector_type(16))) __bf16 v16bf;
typedef __attribute__((ext_vector_type(8)))  float  v8f;

__device__ __forceinline__ float siluf(float x) { return x / (1.0f + __expf(-x)); }

// ---- WMMA bf16 16x16x32 wrapper ----
__device__ __forceinline__ v8f bwmma(v16bf a, v16bf b, v8f c) {
    return __builtin_amdgcn_wmma_f32_16x16x32_bf16(
        /*neg_a=*/false, a, /*neg_b=*/false, b,
        /*c_mod=*/(short)0, c, /*reuse_a=*/false, /*reuse_b=*/false);
}

// A fragment: 16x32 bf16 tile from a row-major [16][stride] buffer (LDS).
// lane l<16: row l, K = k0+0..15 (elems 0..15)
// lane l>=16: row l-16, K = k0+16..31
__device__ __forceinline__ v16bf load_fragA(const __bf16* base, int stride, int k0) {
    int l = threadIdx.x & 31;
    const __bf16* p = base + (l & 15) * stride + k0 + ((l >> 4) << 4);
    union { unsigned int u[8]; v16bf v; } u;
    const unsigned int* q = (const unsigned int*)p;
    u.u[0] = q[0]; u.u[1] = q[1]; u.u[2] = q[2]; u.u[3] = q[3];
    q = (const unsigned int*)(p + 8);
    u.u[4] = q[0]; u.u[5] = q[1]; u.u[6] = q[2]; u.u[7] = q[3];
    return u.v;
}

// B fragment: 32x16 bf16 tile; weights pre-stored transposed as [Nout][K] bf16,
// so lane l reads column n = l&15 (row n of W^T), 16 contiguous K values:
// lanes 0..15 -> K = k0+0..15 ; lanes 16..31 -> K = k0+16..31
__device__ __forceinline__ v16bf load_fragB(const __bf16* baseT, int stride, int k0) {
    int l = threadIdx.x & 31;
    const __bf16* p = baseT + (l & 15) * stride + k0 + ((l >> 4) << 4);
    union { unsigned int u[8]; v16bf v; } u;
    const unsigned int* q = (const unsigned int*)p;
#pragma unroll
    for (int i = 0; i < 8; ++i) u.u[i] = q[i];
    return u.v;
}

// ---------------- K0: weight transpose + f32->bf16 convert ----------------
// src: [K][Nc] f32 row-major ; dst: [Nc][Kpad] bf16 (zero padded for k>=K)
__global__ void convtr_kernel(const float* __restrict__ src, __bf16* __restrict__ dst,
                              int K, int Nc, int Kpad) {
    int idx = blockIdx.x * blockDim.x + threadIdx.x;
    int total = Nc * Kpad;
    if (idx >= total) return;
    int n = idx / Kpad, k = idx % Kpad;
    dst[idx] = (k < K) ? (__bf16)src[k * Nc + n] : (__bf16)0.0f;
}

// ---------------- zero fill ----------------
__global__ void zero_kernel(float* __restrict__ p, int n) {
    int i = blockIdx.x * blockDim.x + threadIdx.x;
    if (i < n) p[i] = 0.0f;
}

// ---------------- K1: node transforms (pre = lin0(nf), x = lin1(normgate(nf))) ----------------
__global__ void __launch_bounds__(256)
node_kernel(const float* __restrict__ node_feat,
            const __bf16* __restrict__ T0s, const float* __restrict__ lin0_bs,
            const __bf16* __restrict__ T0v,
            const __bf16* __restrict__ T1s, const float* __restrict__ lin1_bs,
            const __bf16* __restrict__ T1v,
            __bf16* __restrict__ ps, float* __restrict__ pv,
            float* __restrict__ xs, float* __restrict__ xv) {
    __shared__ __bf16 As[16][128];       // scalar feats (bf16)
    __shared__ __bf16 Ax[16][128];       // silu(scalar feats)
    __shared__ __bf16 Av[3][16][64];     // vector feats per component
    __shared__ __bf16 Agv[3][16][64];    // gated vector feats
    __shared__ float  Vf[16][64][3];     // f32 vector feats for norm

    int tid = threadIdx.x;
    int node0 = blockIdx.x * 16;

    for (int idx = tid; idx < 16 * DFEAT; idx += 256) {
        int r = idx / DFEAT, c = idx % DFEAT;
        float v = node_feat[(node0 + r) * DFEAT + c];
        if (c < NS) {
            As[r][c] = (__bf16)v;
            Ax[r][c] = (__bf16)siluf(v);
        } else {
            int cc = c - NS, ch = cc / 3, m = cc % 3;
            Vf[r][ch][m] = v;
            Av[m][r][ch] = (__bf16)v;
        }
    }
    __syncthreads();
    for (int idx = tid; idx < 16 * NV; idx += 256) {
        int r = idx >> 6, ch = idx & 63;
        float a = Vf[r][ch][0], b = Vf[r][ch][1], c = Vf[r][ch][2];
        float n = sqrtf(a * a + b * b + c * c + EPSLN * EPSLN);
        float g = siluf(n) / n;
        Agv[0][r][ch] = (__bf16)(a * g);
        Agv[1][r][ch] = (__bf16)(b * g);
        Agv[2][r][ch] = (__bf16)(c * g);
    }
    __syncthreads();

    int w = tid >> 5, l = tid & 31;
    int rbase = (l >> 4) << 3, nn = l & 15;

    // scalar GEMMs: ps = s@lin0_ws + b ; xs = silu(s)@lin1_ws + b  (N-tile per wave)
    {
        int n0 = w * 16;
        v8f cs = {}, cx = {};
#pragma unroll
        for (int k0 = 0; k0 < 128; k0 += 32) {
            v16bf a0 = load_fragA(&As[0][0], 128, k0);
            v16bf ax = load_fragA(&Ax[0][0], 128, k0);
            cs = bwmma(a0, load_fragB(T0s + n0 * 128, 128, k0), cs);
            cx = bwmma(ax, load_fragB(T1s + n0 * 128, 128, k0), cx);
        }
        float b0 = lin0_bs[n0 + nn], b1 = lin1_bs[n0 + nn];
#pragma unroll
        for (int j = 0; j < 8; ++j) {
            int row = node0 + rbase + j;
            ps[row * NS + n0 + nn] = (__bf16)(cs[j] + b0);
            xs[row * NS + n0 + nn] = cx[j] + b1;
        }
    }

    // vector GEMMs: 24 tile tasks (pv: 3 comps x 4 Ntiles, xv: same)
    for (int tt = w; tt < 24; tt += 8) {
        int half = tt / 12, rem = tt % 12;
        int m = rem >> 2, vn0 = (rem & 3) * 16;
        const __bf16* A = half ? &Agv[m][0][0] : &Av[m][0][0];
        const __bf16* B = (half ? T1v : T0v) + vn0 * 64;
        v8f c = {};
#pragma unroll
        for (int k0 = 0; k0 < 64; k0 += 32)
            c = bwmma(load_fragA(A, 64, k0), load_fragB(B, 64, k0), c);
        float* ob = half ? xv : pv;
#pragma unroll
        for (int j = 0; j < 8; ++j) {
            int row = node0 + rbase + j;
            ob[row * (3 * NV) + (vn0 + nn) * 3 + m] = c[j];
        }
    }
}

// ---------------- K3: edge MLPs + messages + scatter-add ----------------
__global__ void __launch_bounds__(256)
edge_kernel(const int* __restrict__ recvI, const int* __restrict__ sendI,
            const float* __restrict__ edge_attr, const float* __restrict__ edge_rshs,
            const __bf16* __restrict__ ps, const float* __restrict__ pv,
            const float* __restrict__ xs, const float* __restrict__ xv,
            const __bf16* __restrict__ Tes1, const float* __restrict__ es_b1,
            const __bf16* __restrict__ Tes2, const float* __restrict__ es_b2,
            const __bf16* __restrict__ Ter1, const float* __restrict__ er_b1,
            const __bf16* __restrict__ Ter2, const float* __restrict__ er_b2,
            float* __restrict__ acc_s, float* __restrict__ acc_v) {
    __shared__ __bf16 S0[16][320];      // [ps_recv | ps_send | vdot]
    __shared__ __bf16 Ea[16][32];       // edge_attr padded to K=32
    __shared__ __bf16 H1[16][128];      // silu hidden (s0 path)
    __shared__ __bf16 H2[16][128];      // silu hidden (edge_attr path)
    __shared__ float  Wt[16][384];      // w = w_es * w_er
    __shared__ float  XsJ[16][128];
    __shared__ float  XvJ[16][64][3];
    __shared__ float  Yr[16][4];
    __shared__ int    rcv[16], snd[16];

    int tid = threadIdx.x;
    int e0 = blockIdx.x * 16;
    if (tid < 16) { rcv[tid] = recvI[e0 + tid]; snd[tid] = sendI[e0 + tid]; }
    __syncthreads();

    for (int idx = tid; idx < 16 * 128; idx += 256) {
        int r = idx >> 7, c = idx & 127;
        S0[r][c]       = ps[rcv[r] * NS + c];
        S0[r][128 + c] = ps[snd[r] * NS + c];
        XsJ[r][c]      = xs[snd[r] * NS + c];
    }
    for (int idx = tid; idx < 16 * 64; idx += 256) {
        int r = idx >> 6, c = idx & 63;
        const float* pr = pv + rcv[r] * 192 + c * 3;
        const float* pq = pv + snd[r] * 192 + c * 3;
        S0[r][256 + c] = (__bf16)(pr[0] * pq[0] + pr[1] * pq[1] + pr[2] * pq[2]);
        const float* xp = xv + snd[r] * 192 + c * 3;
        XvJ[r][c][0] = xp[0]; XvJ[r][c][1] = xp[1]; XvJ[r][c][2] = xp[2];
    }
    for (int idx = tid; idx < 16 * 32; idx += 256) {
        int r = idx >> 5, c = idx & 31;
        Ea[r][c] = (c < 20) ? (__bf16)edge_attr[(e0 + r) * 20 + c] : (__bf16)0.0f;
    }
    if (tid < 64) {
        int r = tid >> 2, j = tid & 3;
        Yr[r][j] = edge_rshs[(e0 + r) * 4 + j];
    }
    __syncthreads();

    int w = tid >> 5, l = tid & 31;
    int rbase = (l >> 4) << 3, nn = l & 15;

    // hidden layers: h1 = silu(s0@es_w1+b1), h2 = silu(ea@er_w1+b1)
    {
        int n0 = w * 16;
        v8f c1 = {}, c2 = {};
#pragma unroll
        for (int k0 = 0; k0 < 320; k0 += 32)
            c1 = bwmma(load_fragA(&S0[0][0], 320, k0), load_fragB(Tes1 + n0 * 320, 320, k0), c1);
        c2 = bwmma(load_fragA(&Ea[0][0], 32, 0), load_fragB(Ter1 + n0 * 32, 32, 0), c2);
        float b1 = es_b1[n0 + nn], b2 = er_b1[n0 + nn];
#pragma unroll
        for (int j = 0; j < 8; ++j) {
            H1[rbase + j][n0 + nn] = (__bf16)siluf(c1[j] + b1);
            H2[rbase + j][n0 + nn] = (__bf16)siluf(c2[j] + b2);
        }
    }
    __syncthreads();

    // output layers + elementwise product -> Wt
    for (int tt = w; tt < 24; tt += 8) {
        int n0 = tt * 16;
        v8f ce = {}, cr = {};
#pragma unroll
        for (int k0 = 0; k0 < 128; k0 += 32) {
            ce = bwmma(load_fragA(&H1[0][0], 128, k0), load_fragB(Tes2 + n0 * 128, 128, k0), ce);
            cr = bwmma(load_fragA(&H2[0][0], 128, k0), load_fragB(Ter2 + n0 * 128, 128, k0), cr);
        }
        float be = es_b2[n0 + nn], br = er_b2[n0 + nn];
#pragma unroll
        for (int j = 0; j < 8; ++j)
            Wt[rbase + j][n0 + nn] = (ce[j] + be) * (cr[j] + br);
    }
    __syncthreads();

    // messages + atomic scatter-add (accumulators are L2-resident)
    const float is3 = 0.57735026918962576f;
    for (int idx = tid; idx < 16 * 192; idx += 256) {
        int r = idx / 192, c = idx % 192;
        float ms;
        if (c < 128) {
            ms = XsJ[r][c] * Yr[r][0] * Wt[r][c];
        } else {
            int c2 = c - 128;
            ms = (XvJ[r][c2][0] * Yr[r][1] + XvJ[r][c2][1] * Yr[r][2] +
                  XvJ[r][c2][2] * Yr[r][3]) * Wt[r][320 + c2] * is3;
        }
        atomicAdd(&acc_s[rcv[r] * 192 + c], ms);
    }
    for (int idx = tid; idx < 16 * 192 * 3; idx += 256) {
        int m = idx % 3, rc = idx / 3;
        int r = rc / 192, c = rc % 192;
        float mv;
        if (c < 128) mv = XsJ[r][c] * Yr[r][1 + m] * Wt[r][128 + c];
        else { int c2 = c - 128; mv = XvJ[r][c2][m] * Yr[r][0] * Wt[r][256 + c2]; }
        atomicAdd(&acc_v[rcv[r] * 576 + c * 3 + m], mv);
    }
}

// ---------------- K4: lin2 + norms + residual ----------------
__global__ void __launch_bounds__(256)
out_kernel(const float* __restrict__ node_feat,
           const float* __restrict__ acc_s, const float* __restrict__ acc_v,
           const __bf16* __restrict__ T2s, const float* __restrict__ lin2_bs,
           const __bf16* __restrict__ T2v,
           const float* __restrict__ ln_gs, const float* __restrict__ ln_bs,
           const float* __restrict__ ln_gv, float* __restrict__ out) {
    __shared__ __bf16 Acs[16][192];
    __shared__ __bf16 Acv[3][16][192];
    __shared__ float  OS[16][128];
    __shared__ float  OV[16][64][3];
    __shared__ float  mu[16], rstd[16], vsc[16];

    int tid = threadIdx.x;
    int node0 = blockIdx.x * 16;

    for (int idx = tid; idx < 16 * 192; idx += 256) {
        int r = idx / 192, c = idx % 192;
        Acs[r][c] = (__bf16)acc_s[(node0 + r) * 192 + c];
        const float* av = acc_v + (node0 + r) * 576 + c * 3;
        Acv[0][r][c] = (__bf16)av[0];
        Acv[1][r][c] = (__bf16)av[1];
        Acv[2][r][c] = (__bf16)av[2];
    }
    __syncthreads();

    int w = tid >> 5, l = tid & 31;
    int rbase = (l >> 4) << 3, nn = l & 15;

    {   // os = acc_s @ lin2_ws + b
        int n0 = w * 16;
        v8f c = {};
#pragma unroll
        for (int k0 = 0; k0 < 192; k0 += 32)
            c = bwmma(load_fragA(&Acs[0][0], 192, k0), load_fragB(T2s + n0 * 192, 192, k0), c);
        float b = lin2_bs[n0 + nn];
#pragma unroll
        for (int j = 0; j < 8; ++j) OS[rbase + j][n0 + nn] = c[j] + b;
    }
    // ov = einsum('ncm,cd->ndm'): 3 comps x 4 Ntiles
    for (int tt = w; tt < 12; tt += 8) {
        int m = tt >> 2, n0 = (tt & 3) * 16;
        v8f c = {};
#pragma unroll
        for (int k0 = 0; k0 < 192; k0 += 32)
            c = bwmma(load_fragA(&Acv[m][0][0], 192, k0), load_fragB(T2v + n0 * 192, 192, k0), c);
#pragma unroll
        for (int j = 0; j < 8; ++j) OV[rbase + j][n0 + nn][m] = c[j];
    }
    __syncthreads();

    if (tid < 16) {
        float s = 0.f;
        for (int c = 0; c < 128; ++c) s += OS[tid][c];
        float m_ = s / 128.f, v_ = 0.f;
        for (int c = 0; c < 128; ++c) { float d = OS[tid][c] - m_; v_ += d * d; }
        mu[tid] = m_;
        rstd[tid] = rsqrtf(v_ / 128.f + EPSLN);
        float n2 = 0.f;
        for (int d = 0; d < 64; ++d)
            for (int m = 0; m < 3; ++m) { float x = OV[tid][d][m]; n2 += x * x; }
        vsc[tid] = rsqrtf(n2 / 64.f + EPSLN);
    }
    __syncthreads();

    for (int idx = tid; idx < 16 * 128; idx += 256) {
        int r = idx >> 7, c = idx & 127;
        float v = (OS[r][c] - mu[r]) * rstd[r] * ln_gs[c] + ln_bs[c];
        int g = (node0 + r) * DFEAT + c;
        out[g] = node_feat[g] + v;
    }
    for (int idx = tid; idx < 16 * 192; idx += 256) {
        int r = idx / 192, rem = idx % 192;
        int d = rem / 3, m = rem % 3;
        float v = OV[r][d][m] * vsc[r] * ln_gv[128 + d];
        int g = (node0 + r) * DFEAT + 128 + d * 3 + m;
        out[g] = node_feat[g] + v;
    }
}

// ---------------- host ----------------
extern "C" void kernel_launch(void* const* d_in, const int* in_sizes, int n_in,
                              void* d_out, int out_size, void* d_ws, size_t ws_size,
                              hipStream_t stream) {
    const float* node_feat = (const float*)d_in[0];
    const float* edge_attr = (const float*)d_in[1];
    const float* edge_rshs = (const float*)d_in[2];
    const int*   edge_idx  = (const int*)d_in[3];
    const float* lin0_ws = (const float*)d_in[4];
    const float* lin0_bs = (const float*)d_in[5];
    const float* lin0_wv = (const float*)d_in[6];
    const float* lin1_ws = (const float*)d_in[7];
    const float* lin1_bs = (const float*)d_in[8];
    const float* lin1_wv = (const float*)d_in[9];
    const float* lin2_ws = (const float*)d_in[10];
    const float* lin2_bs = (const float*)d_in[11];
    const float* lin2_wv = (const float*)d_in[12];
    const float* es_w1 = (const float*)d_in[13];
    const float* es_b1 = (const float*)d_in[14];
    const float* es_w2 = (const float*)d_in[15];
    const float* es_b2 = (const float*)d_in[16];
    const float* er_w1 = (const float*)d_in[17];
    const float* er_b1 = (const float*)d_in[18];
    const float* er_w2 = (const float*)d_in[19];
    const float* er_b2 = (const float*)d_in[20];
    const float* ln_gs = (const float*)d_in[21];
    const float* ln_bs = (const float*)d_in[22];
    const float* ln_gv = (const float*)d_in[23];
    float* out = (float*)d_out;

    // workspace carve-up (256B aligned)
    char* wsb = (char*)d_ws;
    size_t off = 0;
    auto carve = [&](size_t bytes) -> void* {
        void* p = wsb + off;
        off += (bytes + 255) & ~(size_t)255;
        return p;
    };
    __bf16* T0s  = (__bf16*)carve(128 * 128 * 2);
    __bf16* T0v  = (__bf16*)carve(64 * 64 * 2);
    __bf16* T1s  = (__bf16*)carve(128 * 128 * 2);
    __bf16* T1v  = (__bf16*)carve(64 * 64 * 2);
    __bf16* T2s  = (__bf16*)carve(128 * 192 * 2);
    __bf16* T2v  = (__bf16*)carve(64 * 192 * 2);
    __bf16* Tes1 = (__bf16*)carve(128 * 320 * 2);
    __bf16* Tes2 = (__bf16*)carve(384 * 128 * 2);
    __bf16* Ter1 = (__bf16*)carve(128 * 32 * 2);
    __bf16* Ter2 = (__bf16*)carve(384 * 128 * 2);
    __bf16* ps   = (__bf16*)carve((size_t)N_NODES * NS * 2);
    float*  pv   = (float*)carve((size_t)N_NODES * 192 * 4);
    float*  xs   = (float*)carve((size_t)N_NODES * NS * 4);
    float*  xv   = (float*)carve((size_t)N_NODES * 192 * 4);
    float*  acc_s = (float*)carve((size_t)N_NODES * 192 * 4);
    float*  acc_v = (float*)carve((size_t)N_NODES * 576 * 4);

    auto conv = [&](const float* src, __bf16* dst, int K, int Nc, int Kpad) {
        int total = Nc * Kpad;
        convtr_kernel<<<(total + 255) / 256, 256, 0, stream>>>(src, dst, K, Nc, Kpad);
    };
    conv(lin0_ws, T0s, 128, 128, 128);
    conv(lin0_wv, T0v, 64, 64, 64);
    conv(lin1_ws, T1s, 128, 128, 128);
    conv(lin1_wv, T1v, 64, 64, 64);
    conv(lin2_ws, T2s, 192, 128, 192);
    conv(lin2_wv, T2v, 192, 64, 192);
    conv(es_w1, Tes1, 320, 128, 320);
    conv(es_w2, Tes2, 128, 384, 128);
    conv(er_w1, Ter1, 20, 128, 32);
    conv(er_w2, Ter2, 128, 384, 128);

    node_kernel<<<N_NODES / 16, 256, 0, stream>>>(
        node_feat, T0s, lin0_bs, T0v, T1s, lin1_bs, T1v, ps, pv, xs, xv);

    int nacc = N_NODES * (192 + 576);
    zero_kernel<<<(nacc + 255) / 256, 256, 0, stream>>>(acc_s, nacc);  // acc_s/acc_v contiguous

    edge_kernel<<<N_EDGES / 16, 256, 0, stream>>>(
        edge_idx, edge_idx + N_EDGES, edge_attr, edge_rshs,
        ps, pv, xs, xv,
        Tes1, es_b1, Tes2, es_b2, Ter1, er_b1, Ter2, er_b2,
        acc_s, acc_v);

    out_kernel<<<N_NODES / 16, 256, 0, stream>>>(
        node_feat, acc_s, acc_v, T2s, lin2_bs, T2v, ln_gs, ln_bs, ln_gv, out);

    (void)in_sizes; (void)n_in; (void)out_size; (void)ws_size;
}
